// AdaptiveAttention_7670811590922
// MI455X (gfx1250) — compile-verified, matrix operand order checked by reading, not run on hardware
//
#include <hip/hip_runtime.h>

typedef __attribute__((ext_vector_type(16))) _Float16 v16h;
typedef __attribute__((ext_vector_type(8)))  _Float16 v8h;
typedef __attribute__((ext_vector_type(8)))  float    v8f;

static constexpr int Bsz  = 4;
static constexpr int Sseq = 2048;
static constexpr int Dd   = 1024;

static constexpr int BM = 128;   // block tile M (4 waves x 32)
static constexpr int BN = 128;   // block tile N (2 waves x 64)
static constexpr int BK = 32;    // k-step (matches WMMA K)
static constexpr int LDS_STRIDE = BK + 8;  // pad to dodge bank conflicts

// -------- WMMA operand gather from an LDS tile laid out [rows][LDS_STRIDE] f16.
// 16-bit A/B operand layout (ISA 7.12.2): lane&15 = row, lane>>4 selects the
// 8-wide K half; halves 0..7 <- K[hi*8 .. +7], halves 8..15 <- K[16+hi*8 .. +7].
__device__ __forceinline__ v16h frag_from_lds(const _Float16* base, int row0) {
  const int lane = threadIdx.x & 31;
  const int m  = lane & 15;
  const int hi = lane >> 4;
  const _Float16* p = base + (row0 + m) * LDS_STRIDE + hi * 8;
  v8h lo = *(const v8h*)(p);
  v8h hh = *(const v8h*)(p + 16);
  v16h r;
#pragma unroll
  for (int i = 0; i < 8; ++i) { r[i] = lo[i]; r[i + 8] = hh[i]; }
  return r;
}

// MODE 0: QKV projection epilogue (bias, split into q/k/v; v stored transposed)
// MODE 1: scores = scale * Q K^T, f16 out, batched via blockIdx.z
// MODE 2: O = P V, f16 out, batched
// MODE 3: out-proj epilogue (bias), f32 out to d_out
template <int MODE>
__global__ __launch_bounds__(256)
void gemm_wmma(const _Float16* __restrict__ Ag, const _Float16* __restrict__ Bg,
               const float* __restrict__ bias, void* __restrict__ outp,
               int M, int N, int K, int lda, int ldb,
               long aBatch, long bBatch, long oBatch, int ldo, float scale,
               _Float16* __restrict__ qp, _Float16* __restrict__ kp,
               _Float16* __restrict__ vp)
{
  __shared__ alignas(16) _Float16 As[2][BM * LDS_STRIDE];   // 2 x 10 KB
  __shared__ alignas(16) _Float16 Bs[2][BN * LDS_STRIDE];   // 2 x 10 KB

  const int tid  = threadIdx.x;
  const int wave = tid >> 5;
  const int wm   = wave >> 1;          // 0..3 -> 32-row strip
  const int wn   = wave & 1;           // 0..1 -> 64-col strip
  const int m0   = blockIdx.y * BM;
  const int n0   = blockIdx.x * BN;
  const int bz   = blockIdx.z;

  const _Float16* A = Ag + (long)bz * aBatch;
  const _Float16* B = Bg + (long)bz * bBatch;

  // staging map: chunk cid in [0,512): row = cid>>2, col = (cid&3)*8
  const int r0 = tid >> 2;             // rows for cid = tid
  const int r1 = (tid + 256) >> 2;     // rows for cid = tid + 256
  const int c0 = (tid & 3) * 8;

  v8f zero = {};
  v8f acc[2][4];
#pragma unroll
  for (int i = 0; i < 2; ++i)
#pragma unroll
    for (int j = 0; j < 4; ++j) acc[i][j] = zero;

  // ---- prologue: stage k=0 into buffer 0
  {
    *(v8h*)(As[0] + r0 * LDS_STRIDE + c0) = *(const v8h*)(A + (long)(m0 + r0) * lda + c0);
    *(v8h*)(As[0] + r1 * LDS_STRIDE + c0) = *(const v8h*)(A + (long)(m0 + r1) * lda + c0);
    *(v8h*)(Bs[0] + r0 * LDS_STRIDE + c0) = *(const v8h*)(B + (long)(n0 + r0) * ldb + c0);
    *(v8h*)(Bs[0] + r1 * LDS_STRIDE + c0) = *(const v8h*)(B + (long)(n0 + r1) * ldb + c0);
  }
  __syncthreads();

  const int nk = K / BK;
  for (int ks = 0; ks < nk; ++ks) {
    const int cur = ks & 1;
    const bool hasNext = (ks + 1) < nk;

    // issue next tile's global loads early (latency hidden under WMMA burst)
    v8h ta0, ta1, tb0, tb1;
    if (hasNext) {
      const int k0 = (ks + 1) * BK;
      ta0 = *(const v8h*)(A + (long)(m0 + r0) * lda + k0 + c0);
      ta1 = *(const v8h*)(A + (long)(m0 + r1) * lda + k0 + c0);
      tb0 = *(const v8h*)(B + (long)(n0 + r0) * ldb + k0 + c0);
      tb1 = *(const v8h*)(B + (long)(n0 + r1) * ldb + k0 + c0);
    }

    // gather all fragments, then back-to-back WMMAs
    v16h af[2], bf[4];
#pragma unroll
    for (int i = 0; i < 2; ++i) af[i] = frag_from_lds(As[cur], wm * 32 + i * 16);
#pragma unroll
    for (int j = 0; j < 4; ++j) bf[j] = frag_from_lds(Bs[cur], wn * 64 + j * 16);
#pragma unroll
    for (int i = 0; i < 2; ++i)
#pragma unroll
      for (int j = 0; j < 4; ++j)
        acc[i][j] = __builtin_amdgcn_wmma_f32_16x16x32_f16(
            false, af[i], false, bf[j], (short)0, acc[i][j], false, false);

    if (hasNext) {
      const int nxt = cur ^ 1;
      *(v8h*)(As[nxt] + r0 * LDS_STRIDE + c0) = ta0;
      *(v8h*)(As[nxt] + r1 * LDS_STRIDE + c0) = ta1;
      *(v8h*)(Bs[nxt] + r0 * LDS_STRIDE + c0) = tb0;
      *(v8h*)(Bs[nxt] + r1 * LDS_STRIDE + c0) = tb1;
    }
    __syncthreads();
  }

  // Epilogue. C/D layout (ISA 7.12.2): lane&15 = N col, VGPR r -> M row
  // r + 8*(lane>>4).
  const int lane  = tid & 31;
  const int nlane = lane & 15;
  const int hi    = lane >> 4;
#pragma unroll
  for (int i = 0; i < 2; ++i) {
#pragma unroll
    for (int j = 0; j < 4; ++j) {
#pragma unroll
      for (int r = 0; r < 8; ++r) {
        const int gm = m0 + wm * 32 + i * 16 + hi * 8 + r;
        const int gn = n0 + wn * 64 + j * 16 + nlane;
        float v = acc[i][j][r];
        if (MODE == 0) {
          v += bias[gn];
          const _Float16 h = (_Float16)v;
          const int b = gm >> 11;                 // gm / Sseq
          const int s = gm & (Sseq - 1);
          if (gn < Dd)          qp[(long)gm * Dd + gn] = h;
          else if (gn < 2 * Dd) kp[(long)gm * Dd + (gn - Dd)] = h;
          else {
            const int vd = gn - 2 * Dd;
            vp[((long)b * Dd + vd) * Sseq + s] = h;   // V stored [b][d][s]
          }
        } else if (MODE == 3) {
          float* o = (float*)outp;
          o[(long)gm * ldo + gn] = v + bias[gn];
        } else {
          _Float16* o = (_Float16*)outp + (long)bz * oBatch;
          o[(long)gm * ldo + gn] = (_Float16)(v * scale);
        }
      }
    }
  }
}

// ---------------- prep kernels ----------------
__global__ void cvt_f32_to_f16(const float* __restrict__ in,
                               _Float16* __restrict__ out, long n) {
  long i = (long)blockIdx.x * blockDim.x + threadIdx.x;
  if (i < n) out[i] = (_Float16)in[i];
}

// in: [K][N] f32 row-major -> out: [N][K] f16 row-major
__global__ void transpose_cvt(const float* __restrict__ in,
                              _Float16* __restrict__ out, int K, int N) {
  long i = (long)blockIdx.x * blockDim.x + threadIdx.x;
  if (i < (long)K * N) {
    const int k = (int)(i / N);
    const int n = (int)(i % N);
    out[(long)n * K + k] = (_Float16)in[i];
  }
}

// ---------------- row softmax over T=2048, in place on f16 ----------------
__global__ __launch_bounds__(256)
void softmax_rows(_Float16* __restrict__ P, int T) {
  __shared__ float red[256];
  const int tid = threadIdx.x;
  _Float16* p = P + (long)blockIdx.x * T;

  float v[8];
  float mx = -3.0e38f;
#pragma unroll
  for (int j = 0; j < 8; ++j) {
    v[j] = (float)p[tid + j * 256];
    mx = fmaxf(mx, v[j]);
  }
  red[tid] = mx; __syncthreads();
  for (int s = 128; s > 0; s >>= 1) {
    if (tid < s) red[tid] = fmaxf(red[tid], red[tid + s]);
    __syncthreads();
  }
  mx = red[0]; __syncthreads();

  float sum = 0.f;
#pragma unroll
  for (int j = 0; j < 8; ++j) { v[j] = __expf(v[j] - mx); sum += v[j]; }
  red[tid] = sum; __syncthreads();
  for (int s = 128; s > 0; s >>= 1) {
    if (tid < s) red[tid] += red[tid + s];
    __syncthreads();
  }
  const float inv = 1.f / red[0];
#pragma unroll
  for (int j = 0; j < 8; ++j) p[tid + j * 256] = (_Float16)(v[j] * inv);
}

// ---------------- host-side orchestration ----------------
extern "C" void kernel_launch(void* const* d_in, const int* in_sizes, int n_in,
                              void* d_out, int out_size, void* d_ws, size_t ws_size,
                              hipStream_t stream)
{
  const float* x    = (const float*)d_in[0];
  const float* Wqkv = (const float*)d_in[1];
  const float* bqkv = (const float*)d_in[2];
  const float* Wout = (const float*)d_in[3];
  const float* bout = (const float*)d_in[4];
  float* out = (float*)d_out;

  char* ws = (char*)d_ws;
  const long MB = 1024l * 1024l;
  _Float16* x_h    = (_Float16*)(ws);             // 16 MB  [8192][1024]
  _Float16* wqkv_t = (_Float16*)(ws + 16 * MB);   //  6 MB  [3072][1024]
  _Float16* wout_t = (_Float16*)(ws + 22 * MB);   //  2 MB  [1024][1024]
  _Float16* q_h    = (_Float16*)(ws + 24 * MB);   // 16 MB  [b][s][d]
  _Float16* k_h    = (_Float16*)(ws + 40 * MB);   // 16 MB  [b][s][d]
  _Float16* v_t    = (_Float16*)(ws + 56 * MB);   // 16 MB  [b][d][s]
  _Float16* probs  = (_Float16*)(ws + 72 * MB);   // 32 MB  [b][s][t]
  _Float16* attno  = (_Float16*)(ws + 104 * MB);  // 16 MB  [8192][1024]

  const long nx = (long)Bsz * Sseq * Dd;                 // 8,388,608
  cvt_f32_to_f16<<<dim3((unsigned)((nx + 255) / 256)), 256, 0, stream>>>(x, x_h, nx);
  transpose_cvt<<<dim3((Dd * 3 * Dd + 255) / 256), 256, 0, stream>>>(Wqkv, wqkv_t, Dd, 3 * Dd);
  transpose_cvt<<<dim3((Dd * Dd + 255) / 256), 256, 0, stream>>>(Wout, wout_t, Dd, Dd);

  // 1) QKV projection: [8192,1024] x [1024,3072] + b
  gemm_wmma<0><<<dim3(3 * Dd / BN, Bsz * Sseq / BM, 1), 256, 0, stream>>>(
      x_h, wqkv_t, bqkv, nullptr,
      Bsz * Sseq, 3 * Dd, Dd, Dd, Dd, 0, 0, 0, 0, 1.0f, q_h, k_h, v_t);

  // 2) scores = scale * Q K^T  (batched over b)
  gemm_wmma<1><<<dim3(Sseq / BN, Sseq / BM, Bsz), 256, 0, stream>>>(
      q_h, k_h, nullptr, probs,
      Sseq, Sseq, Dd, Dd, Dd,
      (long)Sseq * Dd, (long)Sseq * Dd, (long)Sseq * Sseq, Sseq,
      0.03125f /* 1024^-0.5 */, nullptr, nullptr, nullptr);

  // 3) softmax rows (in place, f16)
  softmax_rows<<<dim3(Bsz * Sseq), 256, 0, stream>>>(probs, Sseq);

  // 4) O = P V  (V pre-transposed -> contiguous B operand)
  gemm_wmma<2><<<dim3(Dd / BN, Sseq / BM, Bsz), 256, 0, stream>>>(
      probs, v_t, nullptr, attno,
      Sseq, Dd, Sseq, Sseq, Sseq,
      (long)Sseq * Sseq, (long)Dd * Sseq, (long)Sseq * Dd, Dd,
      1.0f, nullptr, nullptr, nullptr);

  // 5) final = O Wout + bout  (f32 out)
  gemm_wmma<3><<<dim3(Dd / BN, Bsz * Sseq / BM, 1), 256, 0, stream>>>(
      attno, wout_t, bout, out,
      Bsz * Sseq, Dd, Dd, Dd, Dd, 0, 0, 0, Dd,
      1.0f, nullptr, nullptr, nullptr);
}
